// SelfAttention_55533927137653
// MI455X (gfx1250) — compile-verified, matrix operand order checked by reading, not run on hardware
//
#include <hip/hip_runtime.h>
#include <hip/hip_bf16.h>

// ---------------------------------------------------------------------------
// Polar-bias self-attention, fully fused for MI455X (gfx1250, wave32, WMMA).
//   B=8, L=1024, C=384, H=6, D=64, K=2 (radial), N=4 (angular)
// ---------------------------------------------------------------------------

#define Bsz 8
#define Lsz 1024
#define Csz 384
#define Hsz 6
#define Dsz 64
#define EPSV 1e-5f

typedef __attribute__((ext_vector_type(16))) _Float16 v16h;
typedef __attribute__((ext_vector_type(8)))  _Float16 v8h;
typedef __attribute__((ext_vector_type(8)))  float    v8f;

// ---- WMMA wrapper: D = A(16x32 f16) x B(32x16 f16) + C(f32) ---------------
__device__ __forceinline__ v8f wmma16x16x32(v16h a, v16h b, v8f c) {
  return __builtin_amdgcn_wmma_f32_16x16x32_f16(
      /*neg_a=*/false, a, /*neg_b=*/false, b,
      /*c_mod=*/(short)0, c, /*reuse_a=*/false, /*reuse_b=*/false);
}

// ---- Fragment loader ------------------------------------------------------
// A-operand layout (16-bit, 16x32, wave32), per ISA 7.12.2:
//   lane 0-15 : row M=lane,     VGPR0-3 = K 0..7,  VGPR4-7 = K 16..23
//   lane16-31 : row M=lane-16,  VGPR0-3 = K 8..15, VGPR4-7 = K 24..31
// B-operand uses the symmetric (transposed-role) pattern, so the same loader
// serves B when `base` points at a row-major [N][K] tile (i.e. Bt memory).
__device__ __forceinline__ v16h load_tile(const _Float16* __restrict__ base, int ld) {
  const int lane = threadIdx.x & 31;
  const int t  = lane & 15;     // row (A) / column (B)
  const int kh = lane >> 4;     // half-wave K phase
  const _Float16* p = base + t * ld + kh * 8;
  v8h lo = *reinterpret_cast<const v8h*>(p);        // K = kh*8 .. kh*8+7
  v8h hi = *reinterpret_cast<const v8h*>(p + 16);   // K = 16+kh*8 .. +7
  v16h r;
#pragma unroll
  for (int i = 0; i < 8; ++i) { r[i] = lo[i]; r[i + 8] = hi[i]; }
  return r;
}

// ---------------------------------------------------------------------------
// Kernel 1: RMSNorm (fp32 in) -> f16 normalized activations
// one block per token row, 128 threads x 3 elements
// ---------------------------------------------------------------------------
__global__ void k_rmsnorm(const float* __restrict__ x, const float* __restrict__ w,
                          _Float16* __restrict__ xn) {
  const int row = blockIdx.x;
  const float* xr = x + (size_t)row * Csz;
  const int tid = threadIdx.x;
  __shared__ float ssum[4];
  float v0 = xr[tid], v1 = xr[tid + 128], v2 = xr[tid + 256];
  float s = v0 * v0 + v1 * v1 + v2 * v2;
#pragma unroll
  for (int m = 16; m; m >>= 1) s += __shfl_xor(s, m, 32);
  if ((tid & 31) == 0) ssum[tid >> 5] = s;
  __syncthreads();
  float tot = ssum[0] + ssum[1] + ssum[2] + ssum[3];
  float rs = rsqrtf(tot * (1.0f / (float)Csz) + EPSV);
  _Float16* xo = xn + (size_t)row * Csz;
  xo[tid]       = (_Float16)(v0 * rs * w[tid]);
  xo[tid + 128] = (_Float16)(v1 * rs * w[tid + 128]);
  xo[tid + 256] = (_Float16)(v2 * rs * w[tid + 256]);
}

// ---------------------------------------------------------------------------
// Kernel 2: fp32 -> f16 convert (weights)
// ---------------------------------------------------------------------------
__global__ void k_cvt_f16(const float* __restrict__ src, _Float16* __restrict__ dst, int n) {
  int i = blockIdx.x * blockDim.x + threadIdx.x;
  if (i < n) dst[i] = (_Float16)src[i];
}

// ---------------------------------------------------------------------------
// Kernel 3: QKV GEMM  qkv[m][n] = sum_k X[m][k]*Win[n][k] + b_in[n]
// One wave per 16(M) x 64(N) tile: 4 WMMAs per A-fragment load (4x A reuse).
// All 5 fragment loads of a k-step are issued before the first WMMA so the
// scheduler can gate WMMAs on partial LOADcnt and keep VMEM in flight.
// N-tiles are 64-wide = exactly one head of one segment -> wave-uniform
// segment/head decode.  Scatters straight into per-head layouts:
//   Q: [b][h][l][d] (scaled by 1/sqrt(D)),  K: [b][h][l][d],  V: [b][h][d][l]
// ---------------------------------------------------------------------------
__global__ void k_gemm_qkv(const _Float16* __restrict__ X, const _Float16* __restrict__ W,
                           const float* __restrict__ bin,
                           _Float16* __restrict__ Qh, _Float16* __restrict__ Kh,
                           _Float16* __restrict__ Vt) {
  const int m0 = blockIdx.x * 16;
  const int n0 = blockIdx.y * 64;                 // one head-wide column block
  const _Float16* a_base = X + (size_t)m0 * Csz;
  const _Float16* b_base = W + (size_t)n0 * Csz;
  v8f acc[4];
#pragma unroll
  for (int t = 0; t < 4; ++t) acc[t] = (v8f){0.f,0.f,0.f,0.f,0.f,0.f,0.f,0.f};
  for (int k = 0; k < Csz; k += 32) {
    __builtin_prefetch(a_base + k + 64, 0, 1);
    v16h a  = load_tile(a_base + k, Csz);
    v16h b0 = load_tile(b_base + (size_t)0 * 16 * Csz + k, Csz);
    v16h b1 = load_tile(b_base + (size_t)1 * 16 * Csz + k, Csz);
    v16h b2 = load_tile(b_base + (size_t)2 * 16 * Csz + k, Csz);
    v16h b3 = load_tile(b_base + (size_t)3 * 16 * Csz + k, Csz);
    acc[0] = wmma16x16x32(a, b0, acc[0]);
    acc[1] = wmma16x16x32(a, b1, acc[1]);
    acc[2] = wmma16x16x32(a, b2, acc[2]);
    acc[3] = wmma16x16x32(a, b3, acc[3]);
  }
  const int lane = threadIdx.x & 31;
  const int col = lane & 15, g = lane >> 4;
  const int seg = n0 / Csz;                       // 0=Q 1=K 2=V (uniform)
  const int h = (n0 % Csz) >> 6;                  // head (uniform)
  float bv[4];
#pragma unroll
  for (int t = 0; t < 4; ++t) bv[t] = bin[n0 + t * 16 + col];
#pragma unroll
  for (int r = 0; r < 8; ++r) {
    int mrow = m0 + r + 8 * g;
    int bi = mrow >> 10;
    int l  = mrow & (Lsz - 1);
    size_t bh = (size_t)(bi * Hsz + h);
    if (seg == 0) {
#pragma unroll
      for (int t = 0; t < 4; ++t)
        Qh[(bh * Lsz + l) * Dsz + t * 16 + col] = (_Float16)((acc[t][r] + bv[t]) * 0.125f);
    } else if (seg == 1) {
#pragma unroll
      for (int t = 0; t < 4; ++t)
        Kh[(bh * Lsz + l) * Dsz + t * 16 + col] = (_Float16)(acc[t][r] + bv[t]);
    } else {
#pragma unroll
      for (int t = 0; t < 4; ++t)
        Vt[(bh * Dsz + t * 16 + col) * Lsz + l] = (_Float16)(acc[t][r] + bv[t]);
    }
  }
}

// ---------------------------------------------------------------------------
// Polar bias, trig-free: cosT = dx/r, sinT = dy/r, then angle-addition
// recurrences for cos/sin(n*theta), n=0..4;  r-powers r^0..r^2.
// ---------------------------------------------------------------------------
__device__ __forceinline__ float polar_bias(float dx, float dy,
                                            const float* __restrict__ bc,
                                            const float* __restrict__ cc) {
  float r2 = dx * dx + dy * dy;
  float rr = sqrtf(r2);
  float inv = rr > 0.f ? __frcp_rn(rr) : 0.f;
  float c1 = rr > 0.f ? dx * inv : 1.0f;     // atan2(0,0)=0 -> cos=1
  float s1 = dy * inv;                       // rr==0 -> 0
  float c2 = c1 * c1 - s1 * s1, s2 = 2.f * c1 * s1;
  float c3 = c2 * c1 - s2 * s1, s3 = s2 * c1 + c2 * s1;
  float c4 = c3 * c1 - s3 * s1, s4 = s3 * c1 + c3 * s1;
  const float cn[5] = {1.f, c1, c2, c3, c4};
  const float sn[5] = {0.f, s1, s2, s3, s4};
  const float rp[3] = {1.f, rr, r2};
  float acc = 0.f;
#pragma unroll
  for (int k = 0; k < 3; ++k) {
    float t = 0.f;
#pragma unroll
    for (int n = 0; n < 5; ++n) t = fmaf(bc[k * 5 + n], cn[n], fmaf(cc[k * 5 + n], sn[n], t));
    acc = fmaf(rp[k], t, acc);
  }
  return acc;
}

// ---------------------------------------------------------------------------
// Kernel 4: fused flash attention with on-the-fly polar bias.
// One wave per (b,h, 16-query tile); keys processed 32 at a time.
// S via 4 WMMAs (all K-frags loaded first), bias in-register, online softmax
// (half-wave shfl reduce), P relayout through LDS, P*V via 4 WMMAs (V-frags
// loaded before the LDS round-trip so VMEM overlaps the relayout).
// All of Q/K/V for one (b,h) is 384 KB -> whole-problem K/V traffic stays
// resident in MI455X's 192 MB L2; no HBM restreaming, no bias tensor.
// ---------------------------------------------------------------------------
__global__ void k_attn(const _Float16* __restrict__ Qh, const _Float16* __restrict__ Kh,
                       const _Float16* __restrict__ Vt, const float* __restrict__ pos,
                       const float* __restrict__ bco, const float* __restrict__ cco,
                       _Float16* __restrict__ O16) {
  const int qt = blockIdx.x;              // query tile 0..63
  const int bh = blockIdx.y;              // 0..47
  const int b = bh / Hsz, h = bh % Hsz;
  const int lane = threadIdx.x & 31;
  const int col = lane & 15, g = lane >> 4;

  __shared__ __align__(16) _Float16 plds[16 * 32];

  float bc[15], cc[15];
#pragma unroll
  for (int i = 0; i < 15; ++i) { bc[i] = bco[h * 15 + i]; cc[i] = cco[h * 15 + i]; }

  const _Float16* Qp = Qh + ((size_t)bh * Lsz + qt * 16) * Dsz;
  v16h qa0 = load_tile(Qp, Dsz);          // K-dim 0..31
  v16h qa1 = load_tile(Qp + 32, Dsz);     // K-dim 32..63

  float qx[8], qy[8];
#pragma unroll
  for (int r = 0; r < 8; ++r) {
    int l = qt * 16 + r + 8 * g;
    qx[r] = pos[((size_t)b * Lsz + l) * 2 + 0];
    qy[r] = pos[((size_t)b * Lsz + l) * 2 + 1];
  }

  v8f o0 = {0,0,0,0,0,0,0,0}, o1 = o0, o2 = o0, o3 = o0;
  float mrow[8], lrow[8];
#pragma unroll
  for (int r = 0; r < 8; ++r) { mrow[r] = -1e30f; lrow[r] = 0.f; }

  const _Float16* Kbase = Kh + (size_t)bh * Lsz * Dsz;
  const _Float16* Vbase = Vt + (size_t)bh * Dsz * Lsz;
  const float*    pposb = pos + (size_t)b * Lsz * 2;

  for (int j = 0; j < Lsz; j += 32) {
    // ---- scores for keys j..j+31: load all K-frags, then 4 WMMAs ----
    const _Float16* Kp0 = Kbase + (size_t)j * Dsz;
    const _Float16* Kp1 = Kbase + (size_t)(j + 16) * Dsz;
    __builtin_prefetch(Kbase + (size_t)(j + 32) * Dsz, 0, 1);
    v16h kb00 = load_tile(Kp0, Dsz);
    v16h kb01 = load_tile(Kp0 + 32, Dsz);
    v16h kb10 = load_tile(Kp1, Dsz);
    v16h kb11 = load_tile(Kp1 + 32, Dsz);
    v8f s0 = {0,0,0,0,0,0,0,0}, s1 = s0;
    s0 = wmma16x16x32(qa0, kb00, s0);
    s0 = wmma16x16x32(qa1, kb01, s0);
    s1 = wmma16x16x32(qa0, kb10, s1);
    s1 = wmma16x16x32(qa1, kb11, s1);

    // ---- V fragments: issue loads now; they overlap bias + softmax ----
    const _Float16* Vp = Vbase + j;
    v16h vb0 = load_tile(Vp,            Lsz);
    v16h vb1 = load_tile(Vp + 16 * Lsz, Lsz);
    v16h vb2 = load_tile(Vp + 32 * Lsz, Lsz);
    v16h vb3 = load_tile(Vp + 48 * Lsz, Lsz);

    // ---- polar bias (this lane owns key columns j+col and j+16+col) ----
    float kx0 = pposb[(j + col) * 2],      ky0 = pposb[(j + col) * 2 + 1];
    float kx1 = pposb[(j + 16 + col) * 2], ky1 = pposb[(j + 16 + col) * 2 + 1];
#pragma unroll
    for (int r = 0; r < 8; ++r) {
      s0[r] += polar_bias(qx[r] - kx0, qy[r] - ky0, bc, cc);
      s1[r] += polar_bias(qx[r] - kx1, qy[r] - ky1, bc, cc);
    }

    // ---- online softmax (reduce across the 16-lane half-wave) ----
    float scale[8];
#pragma unroll
    for (int r = 0; r < 8; ++r) {
      float mx = fmaxf(s0[r], s1[r]);
#pragma unroll
      for (int msk = 8; msk; msk >>= 1) mx = fmaxf(mx, __shfl_xor(mx, msk, 32));
      float mn = fmaxf(mrow[r], mx);
      float p0 = __expf(s0[r] - mn), p1 = __expf(s1[r] - mn);
      s0[r] = p0; s1[r] = p1;
      float rs = p0 + p1;
#pragma unroll
      for (int msk = 8; msk; msk >>= 1) rs += __shfl_xor(rs, msk, 32);
      float sc = __expf(mrow[r] - mn);
      lrow[r] = lrow[r] * sc + rs;
      mrow[r] = mn;
      scale[r] = sc;
    }
#pragma unroll
    for (int r = 0; r < 8; ++r) {
      o0[r] *= scale[r]; o1[r] *= scale[r]; o2[r] *= scale[r]; o3[r] *= scale[r];
    }

    // ---- relayout P (C-frag, lane=col) -> A-frag (lane=row) via LDS ----
#pragma unroll
    for (int r = 0; r < 8; ++r) {
      int row = r + 8 * g;
      plds[row * 32 + col]      = (_Float16)s0[r];
      plds[row * 32 + 16 + col] = (_Float16)s1[r];
    }
    __syncthreads();
    v16h pa = load_tile(&plds[0], 32);
    __syncthreads();

    // ---- O += P(16x32) x V(32x64) ----
    o0 = wmma16x16x32(pa, vb0, o0);
    o1 = wmma16x16x32(pa, vb1, o1);
    o2 = wmma16x16x32(pa, vb2, o2);
    o3 = wmma16x16x32(pa, vb3, o3);
  }

  // ---- normalize and write O back as f16 in (B, L, C) for the out-proj ----
#pragma unroll
  for (int r = 0; r < 8; ++r) {
    float inv = __frcp_rn(lrow[r]);
    int l = qt * 16 + r + 8 * g;
    size_t base = ((size_t)b * Lsz + l) * Csz + h * Dsz;
    O16[base + col]      = (_Float16)(o0[r] * inv);
    O16[base + 16 + col] = (_Float16)(o1[r] * inv);
    O16[base + 32 + col] = (_Float16)(o2[r] * inv);
    O16[base + 48 + col] = (_Float16)(o3[r] * inv);
  }
}

// ---------------------------------------------------------------------------
// Kernel 5: output projection  out[m][n] = sum_k O[m][k]*Wout[n][k] + b_out[n]
// One wave per 16(M) x 64(N) tile, same load-then-compute scheme as QKV GEMM.
// ---------------------------------------------------------------------------
__global__ void k_gemm_out(const _Float16* __restrict__ O16, const _Float16* __restrict__ Wo,
                           const float* __restrict__ bout, float* __restrict__ out) {
  const int m0 = blockIdx.x * 16;
  const int n0 = blockIdx.y * 64;
  const _Float16* a_base = O16 + (size_t)m0 * Csz;
  const _Float16* b_base = Wo + (size_t)n0 * Csz;
  v8f acc[4];
#pragma unroll
  for (int t = 0; t < 4; ++t) acc[t] = (v8f){0.f,0.f,0.f,0.f,0.f,0.f,0.f,0.f};
  for (int k = 0; k < Csz; k += 32) {
    __builtin_prefetch(a_base + k + 64, 0, 1);
    v16h a  = load_tile(a_base + k, Csz);
    v16h b0 = load_tile(b_base + (size_t)0 * 16 * Csz + k, Csz);
    v16h b1 = load_tile(b_base + (size_t)1 * 16 * Csz + k, Csz);
    v16h b2 = load_tile(b_base + (size_t)2 * 16 * Csz + k, Csz);
    v16h b3 = load_tile(b_base + (size_t)3 * 16 * Csz + k, Csz);
    acc[0] = wmma16x16x32(a, b0, acc[0]);
    acc[1] = wmma16x16x32(a, b1, acc[1]);
    acc[2] = wmma16x16x32(a, b2, acc[2]);
    acc[3] = wmma16x16x32(a, b3, acc[3]);
  }
  const int lane = threadIdx.x & 31;
  const int col = lane & 15, g = lane >> 4;
  float bv[4];
#pragma unroll
  for (int t = 0; t < 4; ++t) bv[t] = bout[n0 + t * 16 + col];
#pragma unroll
  for (int r = 0; r < 8; ++r) {
    int mrow = m0 + r + 8 * g;
#pragma unroll
    for (int t = 0; t < 4; ++t)
      out[(size_t)mrow * Csz + n0 + t * 16 + col] = acc[t][r] + bv[t];
  }
}

// ---------------------------------------------------------------------------
// Host-side launch
// ---------------------------------------------------------------------------
extern "C" void kernel_launch(void* const* d_in, const int* in_sizes, int n_in,
                              void* d_out, int out_size, void* d_ws, size_t ws_size,
                              hipStream_t stream) {
  (void)in_sizes; (void)n_in; (void)out_size; (void)ws_size;
  const float* x      = (const float*)d_in[0];
  const float* pos    = (const float*)d_in[1];
  const float* w_norm = (const float*)d_in[2];
  const float* w_in   = (const float*)d_in[3];
  const float* b_in   = (const float*)d_in[4];
  const float* w_out  = (const float*)d_in[5];
  const float* b_out  = (const float*)d_in[6];
  const float* b_coef = (const float*)d_in[7];
  const float* c_coef = (const float*)d_in[8];
  float* out = (float*)d_out;

  char* ws = (char*)d_ws;
  size_t off = 0;
  auto walloc = [&](size_t bytes) -> void* {
    void* p = ws + off;
    off = (off + bytes + 255) & ~(size_t)255;
    return p;
  };
  const size_t ML = (size_t)Bsz * Lsz;                  // 8192 token rows
  _Float16* X16 = (_Float16*)walloc(ML * Csz * 2);      // normalized activations
  _Float16* W16 = (_Float16*)walloc((size_t)3 * Csz * Csz * 2);
  _Float16* Wo16 = (_Float16*)walloc((size_t)Csz * Csz * 2);
  _Float16* Qh = (_Float16*)walloc(ML * Csz * 2);       // [b][h][l][d]
  _Float16* Kh = (_Float16*)walloc(ML * Csz * 2);       // [b][h][l][d]
  _Float16* Vt = (_Float16*)walloc(ML * Csz * 2);       // [b][h][d][l]
  _Float16* O16 = (_Float16*)walloc(ML * Csz * 2);      // [b][l][c]

  // 1) RMSNorm -> f16
  k_rmsnorm<<<dim3(ML), dim3(128), 0, stream>>>(x, w_norm, X16);

  // 2) weight converts
  {
    int n1 = 3 * Csz * Csz;
    k_cvt_f16<<<dim3((n1 + 255) / 256), dim3(256), 0, stream>>>(w_in, W16, n1);
    int n2 = Csz * Csz;
    k_cvt_f16<<<dim3((n2 + 255) / 256), dim3(256), 0, stream>>>(w_out, Wo16, n2);
  }

  // 3) QKV GEMM (one wave / 16x64 tile): M=8192, N=1152
  k_gemm_qkv<<<dim3(ML / 16, (3 * Csz) / 64), dim3(32), 0, stream>>>(
      X16, W16, b_in, Qh, Kh, Vt);

  // 4) fused flash attention with polar bias: grid (q-tiles, B*H)
  k_attn<<<dim3(Lsz / 16, Bsz * Hsz), dim3(32), 0, stream>>>(
      Qh, Kh, Vt, pos, b_coef, c_coef, O16);

  // 5) output projection (one wave / 16x64 tile): M=8192, N=384
  k_gemm_out<<<dim3(ML / 16, Csz / 64), dim3(32), 0, stream>>>(
      O16, Wo16, b_out, out);
}
